// MultiHeadAttention_2628519985803
// MI455X (gfx1250) — matrix operand order repaired and checked by reading, not compiled, not run on hardware
//
#include <hip/hip_runtime.h>

// ---------------------------------------------------------------------------
// MI455X (gfx1250) multi-head attention, f16 WMMA with f32 accumulation.
//   k1: fused QKV projection (fp32 -> f16; Q pre-scaled by 1/sqrt(dh);
//       Q,K head-split [b][h][s][dh]; V stored TRANSPOSED [b][h][dh][s])
//   k2: flash attention, software-pipelined (K/V frags prefetched in regs,
//       online softmax, P routed C-layout->A-layout through LDS)
//   k3: output projection (f16 ctx x f16 wo + bo -> fp32 out)
// GEMMs use DOUBLE-BUFFERED LDS (one barrier per k-step) and hoist all
// B-fragment ds_loads ahead of the WMMA burst (partial dscnt waits).
// ---------------------------------------------------------------------------

#define D_MODEL 1024
#define HEADS   16
#define DH      64
#define SEQ     2048
#define BATCH   4

typedef _Float16 half8 __attribute__((ext_vector_type(8)));
typedef _Float16 v16h  __attribute__((ext_vector_type(16)));
typedef float    v8f   __attribute__((ext_vector_type(8)));

#define ASTR 40   // LDS row stride (halves): 80B, keeps 8-half runs 16B aligned
#define BSTR 40

static __device__ inline v16h cat16(half8 lo, half8 hi) {
  v16h r;
#pragma unroll
  for (int i = 0; i < 8; ++i) { r[i] = lo[i]; r[8 + i] = hi[i]; }
  return r;
}

static __device__ inline v8f zero8() {
  v8f r;
#pragma unroll
  for (int i = 0; i < 8; ++i) r[i] = 0.f;
  return r;
}

static __device__ inline v8f wmma_f16(v16h a, v16h b, v8f c) {
  // D = A(16x32 f16) * B(32x16 f16) + C(16x16 f32)
  return __builtin_amdgcn_wmma_f32_16x16x32_f16(false, a, false, b,
                                                (short)0, c, false, false);
}

// ---------------------------------------------------------------------------
// Kernel 1: QKV projection.  X(8192x1024 f32) * W(1024x1024 f32) + b.
// blockIdx.z selects {Q,K,V}. Q scaled by 1/sqrt(dh)=0.125.
// Q,K -> f16 [b][h][s][dh];  V -> f16 [b][h][dh][s] (transposed for k2).
// Tile: 128(M) x 64(N) per 256-thread block; double-buffered LDS pipeline.
// ---------------------------------------------------------------------------
__global__ __launch_bounds__(256) void qkv_proj_kernel(
    const float* __restrict__ Xq, const float* __restrict__ Xk,
    const float* __restrict__ Xv,
    const float* __restrict__ Wq, const float* __restrict__ Bq,
    const float* __restrict__ Wk, const float* __restrict__ Bk,
    const float* __restrict__ Wv, const float* __restrict__ Bv,
    _Float16* __restrict__ Qh, _Float16* __restrict__ Kh,
    _Float16* __restrict__ Vh) {
  const int z = blockIdx.z;
  const float* X  = (z == 0) ? Xq : (z == 1) ? Xk : Xv;
  const float* W  = (z == 0) ? Wq : (z == 1) ? Wk : Wv;
  const float* Bb = (z == 0) ? Bq : (z == 1) ? Bk : Bv;
  const float scale = (z == 0) ? 0.125f : 1.0f;

  __shared__ __align__(16) _Float16 As[2][128 * ASTR];  // A tiles (dbl buffer)
  __shared__ __align__(16) _Float16 Wt[2][64 * BSTR];   // W tiles, transposed

  const int tid = threadIdx.x, lane = tid & 31, wid = tid >> 5;
  const int gl = lane >> 4, ln = lane & 15;
  const int m_blk = blockIdx.x * 128;
  const int n_blk = blockIdx.y * 64;

  float4 xr[4], wr[2];  // staging registers (pipeline)
  auto stage_load = [&](int kkv) {
#pragma unroll
    for (int it = 0; it < 4; ++it) {
      int idx = tid + it * 256;              // 1024 float4 units, 128x32 floats
      int row = idx >> 3;
      int c   = (idx & 7) << 2;
      xr[it] = *(const float4*)(X + (size_t)(m_blk + row) * D_MODEL + kkv + c);
    }
#pragma unroll
    for (int it = 0; it < 2; ++it) {
      int idx = tid + it * 256;              // 512 float4 units, 32x64 floats
      int row = idx >> 4;
      int c   = (idx & 15) << 2;
      wr[it] = *(const float4*)(W + (size_t)(kkv + row) * D_MODEL + n_blk + c);
    }
  };
  auto stage_store = [&](int buf) {
    _Float16* Ab = As[buf];
    _Float16* Wb = Wt[buf];
#pragma unroll
    for (int it = 0; it < 4; ++it) {
      int idx = tid + it * 256;
      int row = idx >> 3;
      int c   = (idx & 7) << 2;
      _Float16* dst = Ab + row * ASTR + c;
      dst[0] = (_Float16)(xr[it].x * scale);
      dst[1] = (_Float16)(xr[it].y * scale);
      dst[2] = (_Float16)(xr[it].z * scale);
      dst[3] = (_Float16)(xr[it].w * scale);
    }
#pragma unroll
    for (int it = 0; it < 2; ++it) {
      int idx = tid + it * 256;
      int row = idx >> 4;
      int c   = (idx & 15) << 2;
      Wb[(c + 0) * BSTR + row] = (_Float16)wr[it].x;
      Wb[(c + 1) * BSTR + row] = (_Float16)wr[it].y;
      Wb[(c + 2) * BSTR + row] = (_Float16)wr[it].z;
      Wb[(c + 3) * BSTR + row] = (_Float16)wr[it].w;
    }
  };

  v8f acc[4];
#pragma unroll
  for (int nf = 0; nf < 4; ++nf) acc[nf] = zero8();

  stage_load(0);
  stage_store(0);
  int ibuf = 0;
  for (int kk = 0; kk < D_MODEL; kk += 32, ibuf ^= 1) {
    __syncthreads();                         // tile `ibuf` visible to all
    const bool more = (kk + 32 < D_MODEL);
    if (more) stage_load(kk + 32);           // overlaps ds+wmma below

    // Hoist ALL fragment ds_loads, then burst the 4 WMMAs (partial waits)
    const _Float16* arow = As[ibuf] + (wid * 16 + ln) * ASTR;
    v16h a = cat16(*(const half8*)(arow + gl * 16),
                   *(const half8*)(arow + gl * 16 + 8));
    v16h bf[4];
#pragma unroll
    for (int nf = 0; nf < 4; ++nf) {
      const _Float16* bcol = Wt[ibuf] + (nf * 16 + ln) * BSTR + gl * 16;
      bf[nf] = cat16(*(const half8*)bcol, *(const half8*)(bcol + 8));
    }
#pragma unroll
    for (int nf = 0; nf < 4; ++nf) acc[nf] = wmma_f16(a, bf[nf], acc[nf]);

    if (more) stage_store(ibuf ^ 1);         // other buffer: no barrier needed
  }

  // Epilogue (C layout: lane=col, vgpr r -> row r + 8*gl)
#pragma unroll
  for (int nf = 0; nf < 4; ++nf) {
    int col = n_blk + nf * 16 + ln;
    float bb = Bb[col];
    int h = col >> 6, dh = col & 63;
    int R0 = m_blk + wid * 16 + gl * 8;  // first of 8 consecutive rows (b*S)
    int b_ = R0 >> 11;
    int s0 = R0 & (SEQ - 1);
    if (z == 2) {
      // V transposed [b][h][dh][s]: 8 consecutive s at fixed dh = one half8
      half8 pack;
#pragma unroll
      for (int r = 0; r < 8; ++r) pack[r] = (_Float16)(acc[nf][r] + bb);
      *(half8*)(Vh + ((size_t)(b_ * HEADS + h) * DH + dh) * SEQ + s0) = pack;
    } else {
      _Float16* Out = (z == 0) ? Qh : Kh;
#pragma unroll
      for (int r = 0; r < 8; ++r)
        Out[((size_t)(b_ * HEADS + h) * SEQ + s0 + r) * DH + dh] =
            (_Float16)(acc[nf][r] + bb);
    }
  }
}

// ---------------------------------------------------------------------------
// Kernel 2: flash attention.  Block = 128 threads (4 independent waves);
// each wave owns a 16-query strip and streams 32-key tiles.
// K fragments prefetched one tile ahead; V fragments load contiguously from
// the transposed layout; softmax math hides the memory latency.
// grid = (SEQ/64, BATCH*HEADS)
// ---------------------------------------------------------------------------
__global__ __launch_bounds__(128) void attn_kernel(
    const _Float16* __restrict__ Qh, const _Float16* __restrict__ Kh,
    const _Float16* __restrict__ Vh, const float* __restrict__ maskp,
    _Float16* __restrict__ Ctx) {
  __shared__ __align__(16) _Float16 Pt[4 * 16 * 40];  // per-wave P (16x32)

  const int tid = threadIdx.x, lane = tid & 31, wid = tid >> 5;
  const int gl = lane >> 4, ln = lane & 15;
  const int bh = blockIdx.y, b = bh >> 4, h = bh & 15;
  const int q0 = blockIdx.x * 64 + wid * 16;

  const _Float16* Qb = Qh + (size_t)bh * SEQ * DH;   // [s][dh]
  const _Float16* Kb = Kh + (size_t)bh * SEQ * DH;   // [s][dh]
  const _Float16* Vt = Vh + (size_t)bh * DH * SEQ;   // [dh][s]
  const float* mb = maskp + (size_t)b * SEQ;

  // Q A-fragments held in registers for the whole pass (2 frags cover dh=64)
  v16h qa[2];
  {
    const _Float16* rowp = Qb + (size_t)(q0 + ln) * DH;
#pragma unroll
    for (int kh = 0; kh < 2; ++kh)
      qa[kh] = cat16(*(const half8*)(rowp + kh * 32 + gl * 16),
                     *(const half8*)(rowp + kh * 32 + gl * 16 + 8));
  }

  // K B-fragment loader: kf[g2*2+kh] for key group g2, dh half kh
  auto load_kfrag = [&](int kt_, v16h kf[4]) {
#pragma unroll
    for (int g2 = 0; g2 < 2; ++g2) {
      int skey = kt_ * 32 + g2 * 16 + ln;
      const _Float16* krow = Kb + (size_t)skey * DH + gl * 16;
      kf[g2 * 2 + 0] = cat16(*(const half8*)krow, *(const half8*)(krow + 8));
      kf[g2 * 2 + 1] = cat16(*(const half8*)(krow + 32),
                             *(const half8*)(krow + 40));
    }
  };

  float mrow[8], lrow[8];  // per-lane stats for its 8 rows (row = q0+8*gl+r)
  v8f o[4];
#pragma unroll
  for (int r = 0; r < 8; ++r) { mrow[r] = -3.0e38f; lrow[r] = 0.f; }
#pragma unroll
  for (int nf = 0; nf < 4; ++nf) o[nf] = zero8();
  _Float16* Pw = Pt + wid * 16 * 40;

  v16h kf[4];
  load_kfrag(0, kf);

  for (int kt = 0; kt < SEQ / 32; ++kt) {
    const int k0 = kt * 32;

    // Scores on prefetched K fragments
    v8f sc[2];
    sc[0] = zero8(); sc[1] = zero8();
    sc[0] = wmma_f16(qa[0], kf[0], sc[0]);
    sc[0] = wmma_f16(qa[1], kf[1], sc[0]);
    sc[1] = wmma_f16(qa[0], kf[2], sc[1]);
    sc[1] = wmma_f16(qa[1], kf[3], sc[1]);

    // Issue V fragment loads (contiguous in [dh][s]) — consumed after softmax
    v16h vf[4];
#pragma unroll
    for (int nf = 0; nf < 4; ++nf) {
      const _Float16* vp = Vt + (size_t)(nf * 16 + ln) * SEQ + k0 + gl * 16;
      vf[nf] = cat16(*(const half8*)vp, *(const half8*)(vp + 8));
    }
    // Prefetch next K tile
    v16h kfn[4];
    if (kt + 1 < SEQ / 32) load_kfrag(kt + 1, kfn);
    else { kfn[0] = kf[0]; kfn[1] = kf[1]; kfn[2] = kf[2]; kfn[3] = kf[3]; }

    float mk0 = mb[k0 + ln] * -1.0e9f;
    float mk1 = mb[k0 + 16 + ln] * -1.0e9f;
#pragma unroll
    for (int r = 0; r < 8; ++r) { sc[0][r] += mk0; sc[1][r] += mk1; }

    // Online softmax: row reductions via shfl_xor over the 16-lane half-group
    float al[8];
#pragma unroll
    for (int r = 0; r < 8; ++r) {
      float vmx = fmaxf(sc[0][r], sc[1][r]);
#pragma unroll
      for (int mm = 1; mm < 16; mm <<= 1)
        vmx = fmaxf(vmx, __shfl_xor(vmx, mm, 32));
      float mn   = fmaxf(mrow[r], vmx);
      float aexp = __expf(mrow[r] - mn);
      float p0   = __expf(sc[0][r] - mn);
      float p1   = __expf(sc[1][r] - mn);
      sc[0][r] = p0; sc[1][r] = p1;
      float ps = p0 + p1;
#pragma unroll
      for (int mm = 1; mm < 16; mm <<= 1) ps += __shfl_xor(ps, mm, 32);
      lrow[r] = lrow[r] * aexp + ps;
      mrow[r] = mn;
      al[r] = aexp;
    }
#pragma unroll
    for (int nf = 0; nf < 4; ++nf)
#pragma unroll
      for (int r = 0; r < 8; ++r) o[nf][r] *= al[r];

    // P: C-layout -> LDS -> A-layout (per-wave region; LDS is in-order per
    // wave, one block barrier keeps waves in lockstep)
    {
      int rb = gl * 8;
#pragma unroll
      for (int g2 = 0; g2 < 2; ++g2)
#pragma unroll
        for (int r = 0; r < 8; ++r)
          Pw[(rb + r) * 40 + g2 * 16 + ln] = (_Float16)sc[g2][r];
    }
    __syncthreads();
    v16h pa;
    {
      const _Float16* rowp = Pw + ln * 40;
      pa = cat16(*(const half8*)(rowp + gl * 16),
                 *(const half8*)(rowp + gl * 16 + 8));
    }
#pragma unroll
    for (int nf = 0; nf < 4; ++nf) o[nf] = wmma_f16(pa, vf[nf], o[nf]);

    kf[0] = kfn[0]; kf[1] = kfn[1]; kf[2] = kfn[2]; kf[3] = kfn[3];
  }

  // Normalize and write ctx f16 as [b*S + s][h*64 + dh]
#pragma unroll
  for (int nf = 0; nf < 4; ++nf) {
    int col = h * DH + nf * 16 + ln;
#pragma unroll
    for (int r = 0; r < 8; ++r) {
      int s_ = q0 + gl * 8 + r;
      float inv = 1.0f / lrow[r];
      Ctx[(size_t)(b * SEQ + s_) * D_MODEL + col] = (_Float16)(o[nf][r] * inv);
    }
  }
}

// ---------------------------------------------------------------------------
// Kernel 3: output projection.  ctx(f16 8192x1024) * wo + bo -> fp32 out
// Double-buffered LDS pipeline like k1.
// ---------------------------------------------------------------------------
__global__ __launch_bounds__(256) void out_proj_kernel(
    const _Float16* __restrict__ Ctx, const float* __restrict__ Wo,
    const float* __restrict__ Bo, float* __restrict__ Out) {
  __shared__ __align__(16) _Float16 As[2][128 * ASTR];
  __shared__ __align__(16) _Float16 Wt[2][64 * BSTR];

  const int tid = threadIdx.x, lane = tid & 31, wid = tid >> 5;
  const int gl = lane >> 4, ln = lane & 15;
  const int m_blk = blockIdx.x * 128;
  const int n_blk = blockIdx.y * 64;

  half8 xr[2];
  float4 wr[2];
  auto stage_load = [&](int kkv) {
#pragma unroll
    for (int it = 0; it < 2; ++it) {
      int idx = tid + it * 256;   // 512 half8 units, 128x32 halves
      int row = idx >> 2;
      int c   = (idx & 3) << 3;
      xr[it] = *(const half8*)(Ctx + (size_t)(m_blk + row) * D_MODEL + kkv + c);
    }
#pragma unroll
    for (int it = 0; it < 2; ++it) {
      int idx = tid + it * 256;
      int row = idx >> 4;
      int c   = (idx & 15) << 2;
      wr[it] = *(const float4*)(Wo + (size_t)(kkv + row) * D_MODEL + n_blk + c);
    }
  };
  auto stage_store = [&](int buf) {
    _Float16* Ab = As[buf];
    _Float16* Wb = Wt[buf];
#pragma unroll
    for (int it = 0; it < 2; ++it) {
      int idx = tid + it * 256;
      int row = idx >> 2;
      int c   = (idx & 3) << 3;
      *(half8*)(Ab + row * ASTR + c) = xr[it];
    }
#pragma unroll
    for (int it = 0; it < 2; ++it) {
      int idx = tid + it * 256;
      int row = idx >> 4;
      int c   = (idx & 15) << 2;
      Wb[(c + 0) * BSTR + row] = (_Float16)wr[it].x;
      Wb[(c + 1) * BSTR + row] = (_Float16)wr[it].y;
      Wb[(c + 2) * BSTR + row] = (_Float16)wr[it].z;
      Wb[(c + 3) * BSTR + row] = (_Float16)wr[it].w;
    }
  };

  v8f acc[4];
#pragma unroll
  for (int nf = 0; nf < 4; ++nf) acc[nf] = zero8();

  stage_load(0);
  stage_store(0);
  int ibuf = 0;
  for (int kk = 0; kk < D_MODEL; kk += 32, ibuf ^= 1) {
    __syncthreads();
    const bool more = (kk + 32 < D_MODEL);
    if (more) stage_load(kk + 32);

    const _Float16* arow = As[ibuf] + (wid * 16 + ln) * ASTR;
    v16h a = cat16(*(const half8*)(arow + gl * 16),
                   *(const half8*)(arow + gl * 16 + 8));
    v16h bf[4];
#pragma unroll
    for (int nf = 0; nf < 4; ++nf) {
      const _Float16* bcol = Wt[ibuf] + (nf * 16 + ln) * BSTR + gl * 16;
      bf[nf] = cat16(*(const half8*)bcol, *(const half8*)(bcol + 8));
    }
#pragma unroll
    for (int nf = 0; nf < 4; ++nf) acc[nf] = wmma_f16(a, bf[nf], acc[nf]);

    if (more) stage_store(ibuf ^ 1);
  }

#pragma unroll
  for (int nf = 0; nf < 4; ++nf) {
    int col = n_blk + nf * 16 + ln;
    float bb = Bo[col];
#pragma unroll
    for (int r = 0; r < 8; ++r) {
      int R = m_blk + wid * 16 + gl * 8 + r;
      Out[(size_t)R * D_MODEL + col] = acc[nf][r] + bb;
    }
  }
}

// ---------------------------------------------------------------------------
extern "C" void kernel_launch(void* const* d_in, const int* in_sizes, int n_in,
                              void* d_out, int out_size, void* d_ws,
                              size_t ws_size, hipStream_t stream) {
  const float* Xq    = (const float*)d_in[0];
  const float* Xk    = (const float*)d_in[1];
  const float* Xv    = (const float*)d_in[2];
  const float* maskp = (const float*)d_in[3];
  const float* Wq    = (const float*)d_in[4];
  const float* Bq    = (const float*)d_in[5];
  const float* Wk    = (const float*)d_in[6];
  const float* Bk    = (const float*)d_in[7];
  const float* Wv    = (const float*)d_in[8];
  const float* Bv    = (const float*)d_in[9];
  const float* Wo    = (const float*)d_in[10];
  const float* Bo    = (const float*)d_in[11];
  float* Out = (float*)d_out;

  // Workspace: Q/K/V f16 + ctx f16 = 4 x 16 MiB = 64 MiB
  const size_t nHalf = (size_t)BATCH * HEADS * SEQ * DH;  // 8,388,608
  _Float16* Qh  = (_Float16*)d_ws;
  _Float16* Kh  = Qh + nHalf;
  _Float16* Vh  = Kh + nHalf;   // [b][h][dh][s]
  _Float16* Ctx = Vh + nHalf;

  dim3 gproj((BATCH * SEQ) / 128, D_MODEL / 64, 3);
  qkv_proj_kernel<<<gproj, 256, 0, stream>>>(Xq, Xk, Xv, Wq, Bq, Wk, Bk, Wv, Bv,
                                             Qh, Kh, Vh);

  dim3 gattn(SEQ / 64, BATCH * HEADS);
  attn_kernel<<<gattn, 128, 0, stream>>>(Qh, Kh, Vh, maskp, Ctx);

  dim3 gout((BATCH * SEQ) / 128, D_MODEL / 64);
  out_proj_kernel<<<gout, 256, 0, stream>>>(Ctx, Wo, Bo, Out);

  (void)in_sizes; (void)n_in; (void)out_size; (void)ws_size;
}
